// SelfAttention_v2_11854109737237
// MI455X (gfx1250) — matrix operand order repaired and checked by reading, not compiled, not run on hardware
//
#include <hip/hip_runtime.h>
#include <hip/hip_bf16.h>

typedef __attribute__((ext_vector_type(16))) __bf16 bf16x16;
typedef __attribute__((ext_vector_type(8)))  __bf16 bf16x8;
typedef __attribute__((ext_vector_type(8)))  float  f32x8;

constexpr int DIM  = 1024;   // d_in == d_out
constexpr int NSEQ = 4096;   // sequence length

// ---------------------------------------------------------------------------
// CDNA5 async global->LDS copy (ASYNCcnt path, CDNA5 ISA).
// One 16B element per lane; vdst VGPR holds the wave-relative LDS byte address
// (low 32 bits of a generic pointer to __shared__ memory).
// ---------------------------------------------------------------------------
__device__ __forceinline__ void async_copy_b128(unsigned int lds_off, const void* gptr) {
    asm volatile("global_load_async_to_lds_b128 %0, %1, off"
                 :: "v"(lds_off), "v"((unsigned long long)(uintptr_t)gptr)
                 : "memory");
}
__device__ __forceinline__ void wait_asynccnt0() {
    asm volatile("s_wait_asynccnt 0x0" ::: "memory");
}

// ---------------------------------------------------------------------------
// WMMA operand loaders (wave32; this code's contiguous k order, same in A and B)
// A (16x32, 16-bit): lanes 0-15 -> M=lane, K 0..15;
//                    lanes 16-31 -> M=lane-16, K 16..31
// ---------------------------------------------------------------------------
__device__ __forceinline__ bf16x16 load_a_bf16(const __bf16* src, int ld) {
    const int lane = threadIdx.x & 31;
    const int m    = lane & 15;
    const int kh   = lane >> 4;            // 0 or 1
    const __bf16* p0 = src + m * ld + kh * 16;
    bf16x8 lo = *(const bf16x8*)p0;        // K = kh*16 .. +8
    bf16x8 hi = *(const bf16x8*)(p0 + 8);  // K = kh*16 + 8 .. +8
    bf16x16 a;
#pragma unroll
    for (int i = 0; i < 8; ++i) { a[i] = lo[i]; a[i + 8] = hi[i]; }
    return a;
}

__device__ __forceinline__ bf16x16 load_a_f32(const float* src, int ld) {
    const int lane = threadIdx.x & 31;
    const int m    = lane & 15;
    const int kh   = lane >> 4;
    const float* p0 = src + m * ld + kh * 16;
    f32x8 lo = *(const f32x8*)p0;
    f32x8 hi = *(const f32x8*)(p0 + 8);
    bf16x16 a;
#pragma unroll
    for (int i = 0; i < 8; ++i) { a[i] = (__bf16)lo[i]; a[i + 8] = (__bf16)hi[i]; }
    return a;
}

// B (32x16, 16-bit): lane holds column N=lane&15; lanes 0-15 rows K=0..15,
// lanes 16-31 rows K=16..31 (16 contiguous elements along K).
// src points at &Mat[n0][k0] of a row-major matrix whose ROW is the B-column.
__device__ __forceinline__ bf16x16 load_b_bf16(const __bf16* src, int ld) {
    const int lane = threadIdx.x & 31;
    const int n    = lane & 15;
    const int kg   = lane >> 4;
    return *(const bf16x16*)(src + n * ld + kg * 16);
}

__device__ __forceinline__ bf16x16 load_b_f32(const float* src, int ld) {
    const int lane = threadIdx.x & 31;
    const int n    = lane & 15;
    const int kg   = lane >> 4;
    const float* p = src + n * ld + kg * 16;
    f32x8 lo = *(const f32x8*)p;
    f32x8 hi = *(const f32x8*)(p + 8);
    bf16x16 b;
#pragma unroll
    for (int i = 0; i < 8; ++i) { b[i] = (__bf16)lo[i]; b[i + 8] = (__bf16)hi[i]; }
    return b;
}

// ---------------------------------------------------------------------------
// fp32 -> bf16 streaming conversion (b128 in, b128 out)
// ---------------------------------------------------------------------------
__global__ __launch_bounds__(256)
void cvt_bf16_kernel(const float* __restrict__ src, __bf16* __restrict__ dst, int nvec) {
    const int stride = gridDim.x * blockDim.x;
    const f32x8* s = (const f32x8*)src;
    bf16x8* d = (bf16x8*)dst;
    for (int i = blockIdx.x * blockDim.x + threadIdx.x; i < nvec; i += stride) {
        f32x8 v = s[i];
        bf16x8 o;
#pragma unroll
        for (int j = 0; j < 8; ++j) o[j] = (__bf16)v[j];
        d[i] = o;
    }
}

// ---------------------------------------------------------------------------
// QKV epilogue (shared by both QKV variants): store one wave's 32x128 strip.
// ---------------------------------------------------------------------------
__device__ __forceinline__ void qkv_store(f32x8 (&acc)[2][8], int m0, int nb, int mat,
                                          __bf16* qb, __bf16* kb, __bf16* vt) {
    const int lane = threadIdx.x & 31;
    const int rb = (lane < 16) ? 0 : 8;
    const int cl = lane & 15;
#pragma unroll
    for (int mt = 0; mt < 2; ++mt) {
#pragma unroll
        for (int j = 0; j < 8; ++j) {
            const int col = nb + j * 16 + cl;
            if (mat == 2) {
                bf16x8 v;
#pragma unroll
                for (int i = 0; i < 8; ++i) v[i] = (__bf16)acc[mt][j][i];
                *(bf16x8*)(vt + (size_t)col * NSEQ + m0 + mt * 16 + rb) = v;
            } else {
                __bf16* dst = (mat == 0) ? qb : kb;
#pragma unroll
                for (int i = 0; i < 8; ++i)
                    dst[(m0 + mt * 16 + rb + i) * DIM + col] = (__bf16)acc[mt][j][i];
            }
        }
    }
}

// ---------------------------------------------------------------------------
// Kernel 1 (fast path): QKV projections from pre-converted bf16 operands.
// grid = (NSEQ/32, 3), block = 256 (8 waves).  Wave w: 32x128 strip.
// ---------------------------------------------------------------------------
__global__ __launch_bounds__(256)
void qkv_bf16_kernel(const __bf16* __restrict__ xb,
                     const __bf16* __restrict__ wb,   // [3][DIM][DIM]
                     __bf16* __restrict__ qb,
                     __bf16* __restrict__ kb,
                     __bf16* __restrict__ vt) {
    const int w   = threadIdx.x >> 5;
    const int m0  = blockIdx.x * 32;
    const int nb  = w * 128;
    const int mat = blockIdx.y;
    const __bf16* W = wb + (size_t)mat * DIM * DIM;

    const f32x8 zero = {0.f, 0.f, 0.f, 0.f, 0.f, 0.f, 0.f, 0.f};
    f32x8 acc[2][8];
#pragma unroll
    for (int mt = 0; mt < 2; ++mt)
#pragma unroll
        for (int j = 0; j < 8; ++j) acc[mt][j] = zero;

    for (int kk = 0; kk < DIM; kk += 32) {
        bf16x16 a0 = load_a_bf16(xb + (size_t)m0 * DIM + kk, DIM);
        bf16x16 a1 = load_a_bf16(xb + (size_t)(m0 + 16) * DIM + kk, DIM);
#pragma unroll
        for (int j = 0; j < 8; ++j) {
            bf16x16 b = load_b_bf16(W + (size_t)(nb + j * 16) * DIM + kk, DIM);
            acc[0][j] = __builtin_amdgcn_wmma_f32_16x16x32_bf16(
                false, a0, false, b, (short)0, acc[0][j], false, false);
            acc[1][j] = __builtin_amdgcn_wmma_f32_16x16x32_bf16(
                false, a1, false, b, (short)0, acc[1][j], false, false);
        }
    }
    qkv_store(acc, m0, nb, mat, qb, kb, vt);
}

// ---------------------------------------------------------------------------
// Kernel 1 (fallback path): same tiling, fp32 inputs converted in-register.
// ---------------------------------------------------------------------------
__global__ __launch_bounds__(256)
void qkv_f32_kernel(const float* __restrict__ x,
                    const float* __restrict__ Wq,
                    const float* __restrict__ Wk,
                    const float* __restrict__ Wv,
                    __bf16* __restrict__ qb,
                    __bf16* __restrict__ kb,
                    __bf16* __restrict__ vt) {
    const int w   = threadIdx.x >> 5;
    const int m0  = blockIdx.x * 32;
    const int nb  = w * 128;
    const int mat = blockIdx.y;
    const float* W = (mat == 0) ? Wq : (mat == 1) ? Wk : Wv;

    const f32x8 zero = {0.f, 0.f, 0.f, 0.f, 0.f, 0.f, 0.f, 0.f};
    f32x8 acc[2][8];
#pragma unroll
    for (int mt = 0; mt < 2; ++mt)
#pragma unroll
        for (int j = 0; j < 8; ++j) acc[mt][j] = zero;

    for (int kk = 0; kk < DIM; kk += 32) {
        bf16x16 a0 = load_a_f32(x + (size_t)m0 * DIM + kk, DIM);
        bf16x16 a1 = load_a_f32(x + (size_t)(m0 + 16) * DIM + kk, DIM);
#pragma unroll
        for (int j = 0; j < 8; ++j) {
            bf16x16 b = load_b_f32(W + (size_t)(nb + j * 16) * DIM + kk, DIM);
            acc[0][j] = __builtin_amdgcn_wmma_f32_16x16x32_bf16(
                false, a0, false, b, (short)0, acc[0][j], false, false);
            acc[1][j] = __builtin_amdgcn_wmma_f32_16x16x32_bf16(
                false, a1, false, b, (short)0, acc[1][j], false, false);
        }
    }
    qkv_store(acc, m0, nb, mat, qb, kb, vt);
}

// ---------------------------------------------------------------------------
// Kernel 2: flash attention.  grid = (NSEQ/32, 2), block = 256 (8 waves).
// Q block (32x1024) staged once and K blocks (64x1024) staged per iteration
// into LDS via async global->LDS copies (ASYNCcnt); next-K prefetch is issued
// right after the last K_lds read and overlaps softmax + P@V.
// Row pads (1032 / 72) kill the power-of-two-stride bank conflicts.
// ---------------------------------------------------------------------------
__global__ __launch_bounds__(256)
void attn_kernel(const __bf16* __restrict__ qb,
                 const __bf16* __restrict__ kbm,
                 const __bf16* __restrict__ vt,
                 float* __restrict__ out) {
    constexpr int LDQ = 1032, LDK = 1032, LDP = 72, LDSS = 72;
    __shared__ __bf16 Q_lds[32][LDQ];            // 66 KB
    __shared__ __bf16 K_lds[64][LDK];            // 132 KB
    __shared__ float  S_lds[32][LDSS];           // 9 KB
    __shared__ __bf16 P_lds[32][LDP];            // 4.5 KB
    __shared__ float  m_s[32], l_s[32], sc_s[32];
    __shared__ float  red[32][8];

    const int tid  = threadIdx.x;
    const int w    = tid >> 5;
    const int lane = tid & 31;
    const int row0 = blockIdx.x * 32;
    const int col0 = blockIdx.y * 512 + w * 64;  // this wave's output col slice
    const int tm   = w & 1;                      // S-tile row half
    const int tn   = w >> 1;                     // S-tile col (0..3)
    const int rb   = (lane < 16) ? 0 : 8;
    const int cl   = lane & 15;
    const float sscale = 0.03125f;               // 1/sqrt(1024)

    const unsigned int q_base = (unsigned int)(uintptr_t)&Q_lds[0][0];
    const unsigned int k_base = (unsigned int)(uintptr_t)&K_lds[0][0];

    // ---- async-stage Q block (once) and first K block
    for (int v = tid; v < 32 * 128; v += 256) {         // 32 rows x 128 b128/row
        const int r = v >> 7, c = (v & 127) * 8;
        async_copy_b128(q_base + (unsigned int)(r * LDQ + c) * 2,
                        qb + (size_t)(row0 + r) * DIM + c);
    }
    for (int v = tid; v < 64 * 128; v += 256) {         // 64 rows x 128 b128/row
        const int r = v >> 7, c = (v & 127) * 8;
        async_copy_b128(k_base + (unsigned int)(r * LDK + c) * 2,
                        kbm + (size_t)r * DIM + c);
    }

    const f32x8 zero = {0.f, 0.f, 0.f, 0.f, 0.f, 0.f, 0.f, 0.f};
    f32x8 acc[2][4];
#pragma unroll
    for (int mt = 0; mt < 2; ++mt)
#pragma unroll
        for (int nt = 0; nt < 4; ++nt) acc[mt][nt] = zero;
    if (tid < 32) { m_s[tid] = -3.0e38f; l_s[tid] = 0.f; }

    wait_asynccnt0();
    __syncthreads();

    for (int kp = 0; kp < NSEQ; kp += 64) {
        // ---- S tile from LDS operands: S = Q_rows @ K_rows^T, k = 0..1023
        f32x8 s = zero;
        for (int kk = 0; kk < DIM; kk += 32) {
            bf16x16 a = load_a_bf16(&Q_lds[tm * 16][0] + kk, LDQ);
            bf16x16 b = load_b_bf16(&K_lds[tn * 16][0] + kk, LDK);
            s = __builtin_amdgcn_wmma_f32_16x16x32_bf16(
                false, a, false, b, (short)0, s, false, false);
        }
#pragma unroll
        for (int i = 0; i < 8; ++i)
            S_lds[tm * 16 + rb + i][tn * 16 + cl] = s[i] * sscale;
        __syncthreads();                       // all waves done reading K_lds

        // ---- prefetch next K block; overlaps softmax + P@V below
        if (kp + 64 < NSEQ) {
            for (int v = tid; v < 64 * 128; v += 256) {
                const int r = v >> 7, c = (v & 127) * 8;
                async_copy_b128(k_base + (unsigned int)(r * LDK + c) * 2,
                                kbm + (size_t)(kp + 64 + r) * DIM + c);
            }
        }

        // ---- partial row max (8 threads per row, 8 cols each)
        {
            const int r = tid >> 3, seg = tid & 7;
            float pm = -3.0e38f;
#pragma unroll
            for (int c = 0; c < 8; ++c) pm = fmaxf(pm, S_lds[r][seg * 8 + c]);
            red[r][seg] = pm;
        }
        __syncthreads();
        if (tid < 32) {
            float rm = red[tid][0];
#pragma unroll
            for (int s2 = 1; s2 < 8; ++s2) rm = fmaxf(rm, red[tid][s2]);
            const float mo = m_s[tid];
            const float mn = fmaxf(mo, rm);
            sc_s[tid] = __expf(mo - mn);
            m_s[tid]  = mn;
        }
        __syncthreads();

        // ---- P = exp(S - m), bf16 into LDS, partial sums
        {
            const int r = tid >> 3, seg = tid & 7;
            const float mr = m_s[r];
            float ps = 0.f;
#pragma unroll
            for (int c = 0; c < 8; ++c) {
                const float p = __expf(S_lds[r][seg * 8 + c] - mr);
                P_lds[r][seg * 8 + c] = (__bf16)p;
                ps += p;
            }
            red[r][seg] = ps;
        }
        __syncthreads();
        if (tid < 32) {
            float t = 0.f;
#pragma unroll
            for (int s2 = 0; s2 < 8; ++s2) t += red[tid][s2];
            l_s[tid] = l_s[tid] * sc_s[tid] + t;
        }

        // ---- rescale accumulators, then O += P @ V  (wave's 64-col slice)
#pragma unroll
        for (int mt = 0; mt < 2; ++mt) {
            float f[8];
#pragma unroll
            for (int i = 0; i < 8; ++i) f[i] = sc_s[mt * 16 + rb + i];
#pragma unroll
            for (int nt = 0; nt < 4; ++nt)
#pragma unroll
                for (int i = 0; i < 8; ++i) acc[mt][nt][i] *= f[i];
#pragma unroll
            for (int kk2 = 0; kk2 < 64; kk2 += 32) {
                bf16x16 ap = load_a_bf16(&P_lds[mt * 16][0] + kk2, LDP);
#pragma unroll
                for (int nt = 0; nt < 4; ++nt) {
                    bf16x16 bv = load_b_bf16(
                        vt + (size_t)(col0 + nt * 16) * NSEQ + kp + kk2, NSEQ);
                    acc[mt][nt] = __builtin_amdgcn_wmma_f32_16x16x32_bf16(
                        false, ap, false, bv, (short)0, acc[mt][nt], false, false);
                }
            }
        }
        wait_asynccnt0();                      // next K block landed
        __syncthreads();
    }

    // ---- epilogue: O /= l, fp32 store
#pragma unroll
    for (int mt = 0; mt < 2; ++mt) {
        float inv[8];
#pragma unroll
        for (int i = 0; i < 8; ++i) inv[i] = 1.0f / l_s[mt * 16 + rb + i];
#pragma unroll
        for (int nt = 0; nt < 4; ++nt)
#pragma unroll
            for (int i = 0; i < 8; ++i)
                out[(size_t)(row0 + mt * 16 + rb + i) * DIM + col0 + nt * 16 + cl] =
                    acc[mt][nt][i] * inv[i];
    }
}

// ---------------------------------------------------------------------------
extern "C" void kernel_launch(void* const* d_in, const int* in_sizes, int n_in,
                              void* d_out, int out_size, void* d_ws, size_t ws_size,
                              hipStream_t stream) {
    const float* x  = (const float*)d_in[0];
    const float* Wq = (const float*)d_in[1];
    const float* Wk = (const float*)d_in[2];
    const float* Wv = (const float*)d_in[3];
    float* out = (float*)d_out;

    // workspace layout (bf16 elements):
    //   qb [NSEQ*DIM] | kb [NSEQ*DIM] | vt [DIM*NSEQ] | xb [NSEQ*DIM] | wb [3*DIM*DIM]
    const size_t SZ_QKV = (size_t)3 * NSEQ * DIM;
    const size_t SZ_ALL = SZ_QKV + (size_t)NSEQ * DIM + (size_t)3 * DIM * DIM;
    __bf16* qb = (__bf16*)d_ws;
    __bf16* kb = qb + (size_t)NSEQ * DIM;
    __bf16* vt = kb + (size_t)NSEQ * DIM;

    if (ws_size >= SZ_ALL * sizeof(__bf16)) {
        __bf16* xb = vt + (size_t)DIM * NSEQ;
        __bf16* wb = xb + (size_t)NSEQ * DIM;
        cvt_bf16_kernel<<<dim3(512), dim3(256), 0, stream>>>(x, xb, NSEQ * DIM / 8);
        cvt_bf16_kernel<<<dim3(256), dim3(256), 0, stream>>>(Wq, wb, DIM * DIM / 8);
        cvt_bf16_kernel<<<dim3(256), dim3(256), 0, stream>>>(Wk, wb + (size_t)DIM * DIM, DIM * DIM / 8);
        cvt_bf16_kernel<<<dim3(256), dim3(256), 0, stream>>>(Wv, wb + (size_t)2 * DIM * DIM, DIM * DIM / 8);
        qkv_bf16_kernel<<<dim3(NSEQ / 32, 3), dim3(256), 0, stream>>>(xb, wb, qb, kb, vt);
    } else {
        qkv_f32_kernel<<<dim3(NSEQ / 32, 3), dim3(256), 0, stream>>>(x, Wq, Wk, Wv, qb, kb, vt);
    }
    attn_kernel<<<dim3(NSEQ / 32, 2), dim3(256), 0, stream>>>(qb, kb, vt, out);
}
